// ScaledDotProductAttention_27608049778920
// MI455X (gfx1250) — matrix operand order repaired and checked by reading, not compiled, not run on hardware
//
#include <hip/hip_runtime.h>
#include <hip/hip_bf16.h>

typedef __attribute__((ext_vector_type(16))) _Float16 v16h;
typedef __attribute__((ext_vector_type(8)))  float    v8f;

#define BB 4
#define HH 16
#define SS 2048
#define DD 64
// exp(s * 1/sqrt(64)) == exp2(s * (0.125 * log2(e)))
#define SCALE_LOG2E 0.18033688011112042f

#define WMMA_F32(a, b, c) \
  __builtin_amdgcn_wmma_f32_16x16x32_f16(false, (a), false, (b), (short)0, (c), false, false)

// Raw v_exp_f32 (no denorm guard) -- inputs here are O(+-10), and masked
// lanes are zeroed by `keep` anyway, so the libdevice guard is dead weight.
__device__ __forceinline__ float fast_exp2(float x) {
  return __builtin_amdgcn_exp2f(x);
}

// xor-butterfly add across the 16-lane half that owns each C/D-layout row.
// ds_swizzle group-of-32 pattern: offset = xor_mask<<10 | and_mask(0x1f)
__device__ __forceinline__ float half16_sum(float v) {
  v += __int_as_float(__builtin_amdgcn_ds_swizzle(__float_as_int(v), 0x041F)); // SWAPX1
  v += __int_as_float(__builtin_amdgcn_ds_swizzle(__float_as_int(v), 0x081F)); // SWAPX2
  v += __int_as_float(__builtin_amdgcn_ds_swizzle(__float_as_int(v), 0x101F)); // SWAPX4
  v += __int_as_float(__builtin_amdgcn_ds_swizzle(__float_as_int(v), 0x201F)); // SWAPX8
  return v;
}

__global__ __launch_bounds__(128) void fa_fwd_attn_kernel(
    const float* __restrict__ Q, const float* __restrict__ Km,
    const float* __restrict__ Vm, const unsigned char* __restrict__ mask,
    float* __restrict__ ctx, float* __restrict__ attn)
{
  // Per-wave 16x32 f16 P staging buffer (C/D layout -> A layout transform)
  __shared__ _Float16 pbuf[4][16][32];

  const int lane = threadIdx.x & 31;
  const int wave = threadIdx.x >> 5;
  const int tile = blockIdx.x & 31;      // 32 q-tiles of 64 rows per (b,h)
  const int bh   = blockIdx.x >> 5;     // 0..63
  const int b    = bh / HH;
  const int qbase = tile * 64 + wave * 16;

  const int l16   = lane & 15;
  const int hi    = lane >> 4;          // 0: lanes 0-15, 1: lanes 16-31
  const int koffA = hi * 8;             // A-frag K interleave
  const int kb2   = hi * 16;            // B-frag K split

  const float* Qb = Q  + (size_t)bh * SS * DD;
  const float* Kb = Km + (size_t)bh * SS * DD;
  const float* Vb = Vm + (size_t)bh * SS * DD;
  const unsigned char* mb = mask + (size_t)b * SS;

  // ---- Load Q A-fragments (16 rows x 64 d, two K=32 frags), f32 -> f16 ----
  const float* qrow = Qb + (size_t)(qbase + l16) * DD;
  v16h a0, a1;
#pragma unroll
  for (int j = 0; j < 8; ++j) {
    a0[j]     = (_Float16)qrow[kb2 + j];
    a0[8 + j] = (_Float16)qrow[kb2 + 8 + j];
    a1[j]     = (_Float16)qrow[32 + kb2 + j];
    a1[8 + j] = (_Float16)qrow[40 + kb2 + j];
  }

  // ================= Pass 1: row sums of exp(scale*s), mask -> 0 ==========
  float sum[8];
#pragma unroll
  for (int r = 0; r < 8; ++r) sum[r] = 0.0f;

  for (int kt = 0; kt < SS / 16; ++kt) {
    const int kbase = kt * 16;
    const float* krow = Kb + (size_t)(kbase + l16) * DD;
    v16h b0, b1;
#pragma unroll
    for (int j = 0; j < 16; ++j) {
      b0[j] = (_Float16)krow[kb2 + j];        // B(kd, n) = K[n][kd],  kd 0..31
      b1[j] = (_Float16)krow[32 + kb2 + j];   // kd 32..63
    }
    v8f c;
#pragma unroll
    for (int r = 0; r < 8; ++r) c[r] = 0.0f;
    c = WMMA_F32(a0, b0, c);
    c = WMMA_F32(a1, b1, c);

    const float keep = mb[kbase + l16] ? 0.0f : 1.0f;  // mask==1 -> -1e9 -> exp==0
#pragma unroll
    for (int r = 0; r < 8; ++r)
      sum[r] += keep * fast_exp2(c[r] * SCALE_LOG2E);
  }

  // Reduce across the 16-lane half owning each row; keep reciprocal.
#pragma unroll
  for (int r = 0; r < 8; ++r)
    sum[r] = 1.0f / half16_sum(sum[r]);   // inv row sum for row qbase + r + 8*hi

  // ================= Pass 2: write attn, accumulate context = P @ V =======
  v8f ctxacc[4];
#pragma unroll
  for (int n = 0; n < 4; ++n)
#pragma unroll
    for (int r = 0; r < 8; ++r) ctxacc[n][r] = 0.0f;

  for (int ch = 0; ch < SS / 32; ++ch) {
#pragma unroll
    for (int sub = 0; sub < 2; ++sub) {
      const int kbase = ch * 32 + sub * 16;
      const float* krow = Kb + (size_t)(kbase + l16) * DD;
      v16h b0, b1;
#pragma unroll
      for (int j = 0; j < 16; ++j) {
        b0[j] = (_Float16)krow[kb2 + j];
        b1[j] = (_Float16)krow[32 + kb2 + j];
      }
      v8f c;
#pragma unroll
      for (int r = 0; r < 8; ++r) c[r] = 0.0f;
      c = WMMA_F32(a0, b0, c);
      c = WMMA_F32(a1, b1, c);

      const float keep = mb[kbase + l16] ? 0.0f : 1.0f;
#pragma unroll
      for (int r = 0; r < 8; ++r) {
        const float p = keep * fast_exp2(c[r] * SCALE_LOG2E) * sum[r];
        const int row = qbase + r + hi * 8;
        attn[((size_t)bh * SS + row) * SS + kbase + l16] = p;
        pbuf[wave][r + hi * 8][sub * 16 + l16] = (_Float16)p;
      }
    }

    // Same-wave LDS RAW: ensure P stores land before A-layout reload.
    asm volatile("s_wait_dscnt 0" ::: "memory");

    // Reload P tile (16 q x 32 keys) as an A-fragment.
    v16h pa;
#pragma unroll
    for (int j = 0; j < 8; ++j) {
      pa[j]     = pbuf[wave][l16][kb2 + j];
      pa[8 + j] = pbuf[wave][l16][kb2 + 8 + j];
    }

    const int kb = ch * 32;
#pragma unroll
    for (int n = 0; n < 4; ++n) {
      v16h bv;   // B(kkey, dcol) = V[kb + kkey][16n + dcol]
#pragma unroll
      for (int j = 0; j < 16; ++j)
        bv[j] = (_Float16)Vb[(size_t)(kb + kb2 + j) * DD + n * 16 + l16];
      ctxacc[n] = WMMA_F32(pa, bv, ctxacc[n]);
    }
  }

  // ---- Write context (C/D layout -> row-major [bh, row, d]) ----
#pragma unroll
  for (int n = 0; n < 4; ++n)
#pragma unroll
    for (int r = 0; r < 8; ++r) {
      const int row = qbase + r + hi * 8;
      ctx[((size_t)bh * SS + row) * DD + n * 16 + l16] = ctxacc[n][r];
    }
}

extern "C" void kernel_launch(void* const* d_in, const int* in_sizes, int n_in,
                              void* d_out, int out_size, void* d_ws, size_t ws_size,
                              hipStream_t stream) {
  (void)in_sizes; (void)n_in; (void)out_size; (void)d_ws; (void)ws_size;
  const float* Q = (const float*)d_in[0];
  const float* K = (const float*)d_in[1];
  const float* V = (const float*)d_in[2];
  const unsigned char* mask = (const unsigned char*)d_in[3];

  float* ctx  = (float*)d_out;                                   // [4,16,2048,64]
  float* attn = (float*)d_out + (size_t)BB * HH * SS * DD;       // [4,16,2048,2048]

  // One workgroup = 4 waves = 64 q rows; 32 q-tiles per (b,h); 64 (b,h) pairs.
  dim3 grid(BB * HH * (SS / 64));
  dim3 block(128);
  fa_fwd_attn_kernel<<<grid, block, 0, stream>>>(Q, K, V, mask, ctx, attn);
}
